// MoEContradictionClassifier_67680094650866
// MI455X (gfx1250) — matrix operand order repaired and checked by reading, not compiled, not run on hardware
//
#include <hip/hip_runtime.h>
#include <hip/hip_bf16.h>
#include <math.h>

// ---------------- model dims ----------------
constexpr int B = 16, S = 128, D = 768, H = 12, L = 4, F = 3072;
constexpr int E = 8, R = 16, GH = 512, DH = 64;
constexpr int BS = B * S;
constexpr float LORA_SCALE = 2.0f;   // ALPHA/R = 32/16
constexpr float EPS = 1e-5f;

typedef __attribute__((ext_vector_type(16))) __bf16 v16bf;
typedef __attribute__((ext_vector_type(8)))  __bf16 v8bf;
typedef __attribute__((ext_vector_type(8)))  float  v8f;

// LDS rows padded to 40 bf16 (80 B) to de-cluster banks.
constexpr int PAD = 40;

#define WMMA_BF16(a, b, c) \
  __builtin_amdgcn_wmma_f32_16x16x32_bf16(false, (a), false, (b), (short)0, (c), false, false)

// ---------------- device helpers ----------------
__device__ __forceinline__ float block_sum256(float v) {
  __shared__ float sh[8];
  #pragma unroll
  for (int d = 16; d > 0; d >>= 1) v += __shfl_xor(v, d, 32);
  int lane = threadIdx.x & 31, wv = threadIdx.x >> 5;
  __syncthreads();
  if (lane == 0) sh[wv] = v;
  __syncthreads();
  return sh[0] + sh[1] + sh[2] + sh[3] + sh[4] + sh[5] + sh[6] + sh[7];
}

__device__ __forceinline__ float gelu_exact(float x) {
  return 0.5f * x * (1.0f + erff(x * 0.70710678118654752f));
}

// A fragment: row holds 32 contiguous K. kh picks K{kh*8..+7} then {16+kh*8..+7}.
__device__ __forceinline__ v16bf frag_a(const __bf16* rowp, int kh) {
  const v8bf* p = (const v8bf*)rowp;
  v8bf lo = p[kh], hi = p[2 + kh];
  return __builtin_shufflevector(lo, hi, 0,1,2,3,4,5,6,7,8,9,10,11,12,13,14,15);
}

// B fragment: staged column holds 32 contiguous K. kh picks K{kh*8..+7} then {16+kh*8..+7}.
__device__ __forceinline__ v16bf frag_b(const __bf16* colp, int kh) {
  const v8bf* p = (const v8bf*)colp;
  v8bf lo = p[kh], hi = p[2 + kh];
  return __builtin_shufflevector(lo, hi, 0,1,2,3,4,5,6,7,8,9,10,11,12,13,14,15);
}

// ---------------- elementwise / small kernels ----------------
__global__ void k_f32_to_bf16(const float* __restrict__ in, __bf16* __restrict__ out, int n) {
  int i = blockIdx.x * blockDim.x + threadIdx.x;
  if (i < n) out[i] = (__bf16)in[i];
}

__global__ void k_zero(float* p, int n) {
  int i = blockIdx.x * blockDim.x + threadIdx.x;
  if (i < n) p[i] = 0.0f;
}

__global__ void k_embed_ln(const int* __restrict__ ids, const float* __restrict__ tok,
                           const float* __restrict__ pos, const float* __restrict__ g,
                           const float* __restrict__ bb, float* __restrict__ x,
                           __bf16* __restrict__ xb) {
  int row = blockIdx.x;
  int s = row % S;
  int id = ids[row];
  float loc[3], sum = 0.f;
  #pragma unroll
  for (int i = 0; i < 3; ++i) {
    int d = threadIdx.x + 256 * i;
    float v = tok[(size_t)id * D + d] + pos[(size_t)s * D + d];
    loc[i] = v; sum += v;
  }
  float mean = block_sum256(sum) * (1.0f / D);
  float var = 0.f;
  #pragma unroll
  for (int i = 0; i < 3; ++i) { loc[i] -= mean; var += loc[i] * loc[i]; }
  var = block_sum256(var) * (1.0f / D);
  float inv = rsqrtf(var + EPS);
  #pragma unroll
  for (int i = 0; i < 3; ++i) {
    int d = threadIdx.x + 256 * i;
    float v = loc[i] * inv * g[d] + bb[d];
    x[(size_t)row * D + d] = v;
    xb[(size_t)row * D + d] = (__bf16)v;
  }
}

__global__ void k_add_ln(float* __restrict__ x, __bf16* __restrict__ xb,
                         const float* __restrict__ delta, const float* __restrict__ g,
                         const float* __restrict__ bb) {
  int row = blockIdx.x;
  float loc[3], sum = 0.f;
  #pragma unroll
  for (int i = 0; i < 3; ++i) {
    int d = threadIdx.x + 256 * i;
    float v = x[(size_t)row * D + d] + delta[(size_t)row * D + d];
    loc[i] = v; sum += v;
  }
  float mean = block_sum256(sum) * (1.0f / D);
  float var = 0.f;
  #pragma unroll
  for (int i = 0; i < 3; ++i) { loc[i] -= mean; var += loc[i] * loc[i]; }
  var = block_sum256(var) * (1.0f / D);
  float inv = rsqrtf(var + EPS);
  #pragma unroll
  for (int i = 0; i < 3; ++i) {
    int d = threadIdx.x + 256 * i;
    float v = loc[i] * inv * g[d] + bb[d];
    x[(size_t)row * D + d] = v;
    xb[(size_t)row * D + d] = (__bf16)v;
  }
}

__global__ void k_head_ln_gelu(const float* __restrict__ hh, const float* __restrict__ g,
                               const float* __restrict__ bb, __bf16* __restrict__ out) {
  int row = blockIdx.x;
  float loc[2], sum = 0.f;
  #pragma unroll
  for (int i = 0; i < 2; ++i) {
    int d = threadIdx.x + 256 * i;
    float v = hh[(size_t)row * GH + d];
    loc[i] = v; sum += v;
  }
  float mean = block_sum256(sum) * (1.0f / GH);
  float var = 0.f;
  #pragma unroll
  for (int i = 0; i < 2; ++i) { loc[i] -= mean; var += loc[i] * loc[i]; }
  var = block_sum256(var) * (1.0f / GH);
  float inv = rsqrtf(var + EPS);
  #pragma unroll
  for (int i = 0; i < 2; ++i) {
    int d = threadIdx.x + 256 * i;
    out[(size_t)row * GH + d] = (__bf16)gelu_exact(loc[i] * inv * g[d] + bb[d]);
  }
}

// ---------------- WMMA GEMM: C = A[M,K] @ W[K,N] + bias ----------------
// 256 threads = 8 wave32 in 2x4 grid; block tile 64(M) x 128(N); each wave a
// 32x32 register tile (2x2 accumulators -> 4 WMMAs per K-step); K-step 32;
// double-buffered LDS, one barrier per step.
// EPI: 0 = f32; 1 = f32 + bf16; 2 = gelu->bf16; 3 = bf16
template<int EPI>
__global__ void k_gemm(const __bf16* __restrict__ A, const __bf16* __restrict__ W,
                       const float* __restrict__ bias, float* __restrict__ Cf,
                       __bf16* __restrict__ Cb, int M, int N, int Kd) {
  __shared__ __bf16 As[2][64][PAD];
  __shared__ __bf16 Bs[2][128][PAD];
  int tid = threadIdx.x, lane = tid & 31, wvi = tid >> 5;
  int wr = wvi >> 2, wc = wvi & 3;
  int m0 = blockIdx.y * 64, n0 = blockIdx.x * 128;

  auto stage = [&](int kk, int buf) {
    {  // A: 64 rows x 32 K, one v8bf per thread
      int r = tid >> 2, c8 = (tid & 3) << 3;
      int gm = m0 + r;
      v8bf val = {};
      if (gm < M) val = *(const v8bf*)(A + (size_t)gm * Kd + kk + c8);
      *(v8bf*)&As[buf][r][c8] = val;
    }
    #pragma unroll
    for (int it = 0; it < 2; ++it) {  // W: 32 K x 128 N, transpose-scatter
      int i = tid + it * 256;
      int r = i >> 4, c8 = (i & 15) << 3;
      v8bf wv8 = *(const v8bf*)(W + (size_t)(kk + r) * N + n0 + c8);
      #pragma unroll
      for (int j = 0; j < 8; ++j) Bs[buf][c8 + j][r] = wv8[j];
    }
  };

  v8f acc[2][2] = {};
  int nt = Kd / 32;
  stage(0, 0);
  for (int t = 0; t < nt; ++t) {
    __syncthreads();
    if (t + 2 < nt)
      __builtin_prefetch(W + (size_t)((t + 2) * 32 + (tid >> 4)) * N + n0 + ((tid & 15) << 3), 0, 1);
    if (t + 1 < nt) stage((t + 1) * 32, (t + 1) & 1);
    int kh = lane >> 4, lr = lane & 15;
    v16bf a0 = frag_a(&As[t & 1][wr * 32 + lr][0], kh);
    v16bf a1 = frag_a(&As[t & 1][wr * 32 + 16 + lr][0], kh);
    v16bf b0 = frag_b(&Bs[t & 1][wc * 32 + lr][0], kh);
    v16bf b1 = frag_b(&Bs[t & 1][wc * 32 + 16 + lr][0], kh);
    acc[0][0] = WMMA_BF16(a0, b0, acc[0][0]);
    acc[0][1] = WMMA_BF16(a0, b1, acc[0][1]);
    acc[1][0] = WMMA_BF16(a1, b0, acc[1][0]);
    acc[1][1] = WMMA_BF16(a1, b1, acc[1][1]);
  }
  #pragma unroll
  for (int i = 0; i < 2; ++i) {
    int rb = m0 + wr * 32 + i * 16 + (lane >> 4) * 8;
    #pragma unroll
    for (int j = 0; j < 2; ++j) {
      int col = n0 + wc * 32 + j * 16 + (lane & 15);
      float bv = bias ? bias[col] : 0.0f;
      size_t cix = (size_t)rb * N + col;
      #pragma unroll
      for (int v = 0; v < 8; ++v, cix += (size_t)N) {
        if (rb + v >= M) break;
        float val = acc[i][j][v] + bv;
        if (EPI == 0) Cf[cix] = val;
        else if (EPI == 1) { Cf[cix] = val; Cb[cix] = (__bf16)val; }
        else if (EPI == 2) Cb[cix] = (__bf16)gelu_exact(val);
        else Cb[cix] = (__bf16)val;
      }
    }
  }
}

// ---------------- attention ----------------
// scores = (Q @ K^T)/8 + maskbias.  64x128 block tile; grid(S/128, S/64, B*H).
// K-head rows are K-major -> B staging is a straight b128 copy.
__global__ void k_attn_scores(const __bf16* __restrict__ q, const __bf16* __restrict__ k,
                              const int* __restrict__ mask, float* __restrict__ sc) {
  __shared__ __bf16 As[2][64][PAD];
  __shared__ __bf16 Bs[2][128][PAD];
  int tid = threadIdx.x, lane = tid & 31, wvi = tid >> 5;
  int wr = wvi >> 2, wc = wvi & 3;
  int bh = blockIdx.z, b = bh / H, h = bh % H;
  int m0 = blockIdx.y * 64, n0 = blockIdx.x * 128;

  auto stage = [&](int kk, int buf) {
    {
      int r = tid >> 2, c8 = (tid & 3) << 3;
      *(v8bf*)&As[buf][r][c8] =
          *(const v8bf*)(q + (size_t)(b * S + m0 + r) * D + h * DH + kk + c8);
    }
    #pragma unroll
    for (int it = 0; it < 2; ++it) {      // 128 key rows x 32 K, b128 copies
      int i = tid + it * 256;
      int c = i >> 2, r8 = (i & 3) << 3;
      *(v8bf*)&Bs[buf][c][r8] =
          *(const v8bf*)(k + (size_t)(b * S + n0 + c) * D + h * DH + kk + r8);
    }
  };

  v8f acc[2][2] = {};
  stage(0, 0);
  #pragma unroll
  for (int t = 0; t < DH / 32; ++t) {
    __syncthreads();
    if (t + 1 < DH / 32) stage((t + 1) * 32, (t + 1) & 1);
    int kh = lane >> 4, lr = lane & 15;
    v16bf a0 = frag_a(&As[t & 1][wr * 32 + lr][0], kh);
    v16bf a1 = frag_a(&As[t & 1][wr * 32 + 16 + lr][0], kh);
    v16bf b0 = frag_b(&Bs[t & 1][wc * 32 + lr][0], kh);
    v16bf b1 = frag_b(&Bs[t & 1][wc * 32 + 16 + lr][0], kh);
    acc[0][0] = WMMA_BF16(a0, b0, acc[0][0]);
    acc[0][1] = WMMA_BF16(a0, b1, acc[0][1]);
    acc[1][0] = WMMA_BF16(a1, b0, acc[1][0]);
    acc[1][1] = WMMA_BF16(a1, b1, acc[1][1]);
  }
  #pragma unroll
  for (int i = 0; i < 2; ++i) {
    int rb = m0 + wr * 32 + i * 16 + (lane >> 4) * 8;
    #pragma unroll
    for (int j = 0; j < 2; ++j) {
      int col = n0 + wc * 32 + j * 16 + (lane & 15);
      float mb = mask[b * S + col] ? 0.0f : -1e9f;
      size_t cix = ((size_t)bh * S + rb) * S + col;
      #pragma unroll
      for (int v = 0; v < 8; ++v, cix += (size_t)S)
        sc[cix] = acc[i][j][v] * 0.125f + mb;
    }
  }
}

// row softmax over S=128; one wave per row, 8 rows per block.
__global__ void k_softmax(const float* __restrict__ sc, __bf16* __restrict__ pb) {
  int lane = threadIdx.x & 31, wv = threadIdx.x >> 5;
  size_t row = (size_t)blockIdx.x * 8 + wv;
  const float* r = sc + row * S;
  float v[4];
  #pragma unroll
  for (int i = 0; i < 4; ++i) v[i] = r[lane + 32 * i];
  float m = fmaxf(fmaxf(v[0], v[1]), fmaxf(v[2], v[3]));
  #pragma unroll
  for (int d = 16; d > 0; d >>= 1) m = fmaxf(m, __shfl_xor(m, d, 32));
  float s = 0.f;
  #pragma unroll
  for (int i = 0; i < 4; ++i) { v[i] = __expf(v[i] - m); s += v[i]; }
  #pragma unroll
  for (int d = 16; d > 0; d >>= 1) s += __shfl_xor(s, d, 32);
  float inv = 1.0f / s;
  #pragma unroll
  for (int i = 0; i < 4; ++i) pb[row * S + lane + 32 * i] = (__bf16)(v[i] * inv);
}

// O = P @ V_head -> bf16.  32x64 tile (N=DH=64); grid(1, S/32, B*H).
__global__ void k_attn_out(const __bf16* __restrict__ p, const __bf16* __restrict__ v,
                           __bf16* __restrict__ ob) {
  __shared__ __bf16 As[2][32][PAD];
  __shared__ __bf16 Bs[2][64][PAD];
  int tid = threadIdx.x, lane = tid & 31, wvi = tid >> 5;
  int wm = wvi >> 2, wn = wvi & 3;
  int bh = blockIdx.z, b = bh / H, h = bh % H;
  int m0 = blockIdx.y * 32;

  auto stage = [&](int kk, int buf) {
    if (tid < 128) {
      int r = tid >> 2, c8 = (tid & 3) << 3;
      *(v8bf*)&As[buf][r][c8] =
          *(const v8bf*)(p + ((size_t)bh * S + m0 + r) * S + kk + c8);
    }
    int r = tid >> 3, c8 = (tid & 7) << 3;   // V: 32 K x 64 DH transpose-scatter
    v8bf vv8 = *(const v8bf*)(v + (size_t)(b * S + kk + r) * D + h * DH + c8);
    #pragma unroll
    for (int j = 0; j < 8; ++j) Bs[buf][c8 + j][r] = vv8[j];
  };

  v8f acc = {};
  stage(0, 0);
  #pragma unroll
  for (int t = 0; t < S / 32; ++t) {
    __syncthreads();
    if (t + 1 < S / 32) stage((t + 1) * 32, (t + 1) & 1);
    int kh = lane >> 4;
    v16bf af = frag_a(&As[t & 1][wm * 16 + (lane & 15)][0], kh);
    v16bf bfv = frag_b(&Bs[t & 1][wn * 16 + (lane & 15)][0], kh);
    acc = WMMA_BF16(af, bfv, acc);
  }
  int col = wn * 16 + (lane & 15);
  int rb = m0 + wm * 16 + (lane >> 4) * 8;
  size_t cix = (size_t)(b * S + rb) * D + h * DH + col;
  #pragma unroll
  for (int vv = 0; vv < 8; ++vv, cix += (size_t)D)
    ob[cix] = (__bf16)acc[vv];
}

// ---------------- LoRA (rank 16; VALU, bandwidth-trivial) ----------------
__global__ void k_lora_down(const __bf16* __restrict__ x, const float* __restrict__ A,
                            float* __restrict__ tmp, int M) {
  int idx = blockIdx.x * blockDim.x + threadIdx.x;
  if (idx >= M * R) return;
  int m = idx / R, r = idx % R;
  const __bf16* xr = x + (size_t)m * D;
  const float* ar = A + (size_t)r * D;
  float acc = 0.f;
  for (int d = 0; d < D; ++d) acc += (float)xr[d] * ar[d];
  tmp[idx] = acc;
}

__global__ void k_lora_up_add(const float* __restrict__ tmp, const float* __restrict__ Bm,
                              float* __restrict__ out, __bf16* __restrict__ outb,
                              int M, int N) {
  int idx = blockIdx.x * blockDim.x + threadIdx.x;
  if (idx >= M * N) return;
  int m = idx / N, n = idx % N;
  float acc = 0.f;
  #pragma unroll
  for (int r = 0; r < R; ++r) acc += tmp[m * R + r] * Bm[(size_t)n * R + r];
  float val = out[idx] + LORA_SCALE * acc;
  out[idx] = val;
  outb[idx] = (__bf16)val;
}

// ---------------- head tails / gating ----------------
__global__ void k_gather_cls(const float* __restrict__ x, __bf16* __restrict__ cls) {
  int i = blockIdx.x * blockDim.x + threadIdx.x;
  if (i >= B * D) return;
  int b = i / D, d = i % D;
  cls[i] = (__bf16)x[(size_t)(b * S) * D + d];
}

__global__ void k_head_out(const __bf16* __restrict__ hh, const float* __restrict__ W2,
                           const float* __restrict__ b2, float* __restrict__ out, int N) {
  int idx = blockIdx.x * blockDim.x + threadIdx.x;
  if (idx >= B * N) return;
  int b = idx / N, n = idx % N;
  float acc = b2[n];
  for (int j = 0; j < GH; ++j) acc += (float)hh[(size_t)b * GH + j] * W2[(size_t)j * N + n];
  out[idx] = acc;
}

__global__ void k_gate_topk(const float* __restrict__ logits, float* __restrict__ probs_out,
                            float* __restrict__ w) {
  int b = threadIdx.x;
  if (b >= B) return;
  float p[E];
  float m = -1e30f;
  for (int e = 0; e < E; ++e) m = fmaxf(m, logits[b * E + e]);
  float s = 0.f;
  for (int e = 0; e < E; ++e) { p[e] = __expf(logits[b * E + e] - m); s += p[e]; }
  for (int e = 0; e < E; ++e) { p[e] /= s; probs_out[b * E + e] = p[e]; }
  int i0 = 0;
  for (int e = 1; e < E; ++e) if (p[e] > p[i0]) i0 = e;
  int i1 = (i0 == 0) ? 1 : 0;
  for (int e = 0; e < E; ++e) if (e != i0 && p[e] > p[i1]) i1 = e;
  float t = p[i0] + p[i1];
  for (int e = 0; e < E; ++e)
    w[b * E + e] = (e == i0) ? p[i0] / t : (e == i1) ? p[i1] / t : 0.0f;
}

__global__ void k_accum_cls(const float* __restrict__ x, const float* __restrict__ w,
                            int e, float* __restrict__ comb) {
  int i = blockIdx.x * blockDim.x + threadIdx.x;
  if (i >= B * D) return;
  int b = i / D, d = i % D;
  comb[i] += w[b * E + e] * x[(size_t)(b * S) * D + d];
}

// ---------------- host side ----------------
struct WSP {
  __bf16 *Wq, *Wk, *Wv, *Wo, *W1, *W2, *gW1, *cW1;
  float *x; __bf16 *xb;
  float *q, *v; __bf16 *qb, *kb, *vb;
  float *ltmp;
  float *sc; __bf16 *pb;
  __bf16 *ob;
  float *proj; __bf16 *h1b;
  float *f2;
  __bf16 *clsb;
  float *hh; __bf16 *hhb;
  float *glog, *w, *comb; __bf16 *combb;
};

struct Ctx {
  const int *ids, *mask;
  const float *tok, *pos, *ln0g, *ln0b;
  const float *bq, *bk, *bvv, *bo, *ln1g, *ln1b, *b1, *b2, *ln2g, *ln2b;
  WSP ws;
  hipStream_t st;
};

static void encoder_pass(const Ctx& C, const float* Aq, const float* Bq,
                         const float* Av, const float* Bv) {
  hipStream_t st = C.st;
  const WSP& w = C.ws;
  k_embed_ln<<<BS, 256, 0, st>>>(C.ids, C.tok, C.pos, C.ln0g, C.ln0b, w.x, w.xb);
  const int n = BS * D;
  const dim3 gD(D / 128, BS / 64), gF(F / 128, BS / 64);
  for (int l = 0; l < L; ++l) {
    size_t wo = (size_t)l * D * D;
    // Q = x@Wq + bq (+LoRA, fused bf16 out)
    k_gemm<0><<<gD, 256, 0, st>>>(w.xb, w.Wq + wo, C.bq + l * D, w.q, nullptr, BS, D, D);
    k_lora_down<<<(BS * R + 255) / 256, 256, 0, st>>>(w.xb, Aq + (size_t)l * R * D, w.ltmp, BS);
    k_lora_up_add<<<(n + 255) / 256, 256, 0, st>>>(w.ltmp, Bq + (size_t)l * D * R, w.q, w.qb, BS, D);
    // K = x@Wk + bk -> bf16 directly
    k_gemm<3><<<gD, 256, 0, st>>>(w.xb, w.Wk + wo, C.bk + l * D, nullptr, w.kb, BS, D, D);
    // V = x@Wv + bv (+LoRA)
    k_gemm<0><<<gD, 256, 0, st>>>(w.xb, w.Wv + wo, C.bvv + l * D, w.v, nullptr, BS, D, D);
    k_lora_down<<<(BS * R + 255) / 256, 256, 0, st>>>(w.xb, Av + (size_t)l * R * D, w.ltmp, BS);
    k_lora_up_add<<<(n + 255) / 256, 256, 0, st>>>(w.ltmp, Bv + (size_t)l * D * R, w.v, w.vb, BS, D);
    // attention
    k_attn_scores<<<dim3(S / 128, S / 64, B * H), 256, 0, st>>>(w.qb, w.kb, C.mask, w.sc);
    k_softmax<<<(B * H * S) / 8, 256, 0, st>>>(w.sc, w.pb);
    k_attn_out<<<dim3(1, S / 32, B * H), 256, 0, st>>>(w.pb, w.vb, w.ob);
    k_gemm<0><<<gD, 256, 0, st>>>(w.ob, w.Wo + wo, C.bo + l * D, w.proj, nullptr, BS, D, D);
    k_add_ln<<<BS, 256, 0, st>>>(w.x, w.xb, w.proj, C.ln1g + l * D, C.ln1b + l * D);
    // FFN: W1 with fused GELU->bf16; W2 plain f32
    k_gemm<2><<<gF, 256, 0, st>>>(w.xb, w.W1 + (size_t)l * D * F, C.b1 + l * F, nullptr, w.h1b, BS, F, D);
    k_gemm<0><<<gD, 256, 0, st>>>(w.h1b, w.W2 + (size_t)l * F * D, C.b2 + l * D, w.f2, nullptr, BS, D, F);
    k_add_ln<<<BS, 256, 0, st>>>(w.x, w.xb, w.f2, C.ln2g + l * D, C.ln2b + l * D);
  }
}

static void mlp_head_run(const Ctx& C, const __bf16* W1b, const float* b1,
                         const float* g, const float* bb, const float* W2,
                         const float* b2, const __bf16* in, float* logits, int NOUT) {
  hipStream_t st = C.st;
  k_gemm<0><<<dim3(GH / 128, 1), 256, 0, st>>>(in, W1b, b1, C.ws.hh, nullptr, B, GH, D);
  k_head_ln_gelu<<<B, 256, 0, st>>>(C.ws.hh, g, bb, C.ws.hhb);
  k_head_out<<<(B * NOUT + 31) / 32, 32, 0, st>>>(C.ws.hhb, W2, b2, logits, NOUT);
}

extern "C" void kernel_launch(void* const* d_in, const int* in_sizes, int n_in,
                              void* d_out, int out_size, void* d_ws, size_t ws_size,
                              hipStream_t stream) {
  (void)in_sizes; (void)n_in; (void)out_size; (void)ws_size;
  Ctx C;
  C.st = stream;
  C.ids  = (const int*)d_in[0];
  C.mask = (const int*)d_in[1];
  C.tok  = (const float*)d_in[2];
  C.pos  = (const float*)d_in[3];
  C.ln0g = (const float*)d_in[4];
  C.ln0b = (const float*)d_in[5];
  const float* Wq_f = (const float*)d_in[6];  C.bq  = (const float*)d_in[7];
  const float* Wk_f = (const float*)d_in[8];  C.bk  = (const float*)d_in[9];
  const float* Wv_f = (const float*)d_in[10]; C.bvv = (const float*)d_in[11];
  const float* Wo_f = (const float*)d_in[12]; C.bo  = (const float*)d_in[13];
  C.ln1g = (const float*)d_in[14]; C.ln1b = (const float*)d_in[15];
  const float* W1_f = (const float*)d_in[16]; C.b1 = (const float*)d_in[17];
  const float* W2_f = (const float*)d_in[18]; C.b2 = (const float*)d_in[19];
  C.ln2g = (const float*)d_in[20]; C.ln2b = (const float*)d_in[21];
  const float* gAq = (const float*)d_in[22];
  const float* gBq = (const float*)d_in[23];
  const float* gAv = (const float*)d_in[24];
  const float* gBv = (const float*)d_in[25];
  const float* eAq = (const float*)d_in[26];
  const float* eBq = (const float*)d_in[27];
  const float* eAv = (const float*)d_in[28];
  const float* eBv = (const float*)d_in[29];
  const float* gmW1 = (const float*)d_in[30]; const float* gmb1 = (const float*)d_in[31];
  const float* gmg  = (const float*)d_in[32]; const float* gmb  = (const float*)d_in[33];
  const float* gmW2 = (const float*)d_in[34]; const float* gmb2 = (const float*)d_in[35];
  const float* cmW1 = (const float*)d_in[36]; const float* cmb1 = (const float*)d_in[37];
  const float* cmg  = (const float*)d_in[38]; const float* cmb  = (const float*)d_in[39];
  const float* cmW2 = (const float*)d_in[40]; const float* cmb2 = (const float*)d_in[41];

  char* base = (char*)d_ws;
  size_t off = 0;
  auto alloc = [&](size_t bytes) {
    char* p = base + off;
    off = (off + bytes + 255) & ~(size_t)255;
    return p;
  };
  WSP& w = C.ws;
  w.Wq  = (__bf16*)alloc((size_t)L * D * D * 2);
  w.Wk  = (__bf16*)alloc((size_t)L * D * D * 2);
  w.Wv  = (__bf16*)alloc((size_t)L * D * D * 2);
  w.Wo  = (__bf16*)alloc((size_t)L * D * D * 2);
  w.W1  = (__bf16*)alloc((size_t)L * D * F * 2);
  w.W2  = (__bf16*)alloc((size_t)L * F * D * 2);
  w.gW1 = (__bf16*)alloc((size_t)D * GH * 2);
  w.cW1 = (__bf16*)alloc((size_t)D * GH * 2);
  w.x   = (float*)alloc((size_t)BS * D * 4);
  w.xb  = (__bf16*)alloc((size_t)BS * D * 2);
  w.q   = (float*)alloc((size_t)BS * D * 4);
  w.v   = (float*)alloc((size_t)BS * D * 4);
  w.qb  = (__bf16*)alloc((size_t)BS * D * 2);
  w.kb  = (__bf16*)alloc((size_t)BS * D * 2);
  w.vb  = (__bf16*)alloc((size_t)BS * D * 2);
  w.ltmp = (float*)alloc((size_t)BS * R * 4);
  w.sc  = (float*)alloc((size_t)B * H * S * S * 4);
  w.pb  = (__bf16*)alloc((size_t)B * H * S * S * 2);
  w.ob  = (__bf16*)alloc((size_t)BS * D * 2);
  w.proj = (float*)alloc((size_t)BS * D * 4);
  w.h1b = (__bf16*)alloc((size_t)BS * F * 2);
  w.f2  = (float*)alloc((size_t)BS * D * 4);
  w.clsb = (__bf16*)alloc((size_t)B * D * 2);
  w.hh  = (float*)alloc((size_t)B * GH * 4);
  w.hhb = (__bf16*)alloc((size_t)B * GH * 2);
  w.glog = (float*)alloc((size_t)B * E * 4);
  w.w   = (float*)alloc((size_t)B * E * 4);
  w.comb = (float*)alloc((size_t)B * D * 4);
  w.combb = (__bf16*)alloc((size_t)B * D * 2);

  auto cv = [&](const float* src, __bf16* dst, size_t n) {
    k_f32_to_bf16<<<(int)((n + 255) / 256), 256, 0, stream>>>(src, dst, (int)n);
  };
  cv(Wq_f, w.Wq, (size_t)L * D * D);
  cv(Wk_f, w.Wk, (size_t)L * D * D);
  cv(Wv_f, w.Wv, (size_t)L * D * D);
  cv(Wo_f, w.Wo, (size_t)L * D * D);
  cv(W1_f, w.W1, (size_t)L * D * F);
  cv(W2_f, w.W2, (size_t)L * F * D);
  cv(gmW1, w.gW1, (size_t)D * GH);
  cv(cmW1, w.cW1, (size_t)D * GH);

  float* out = (float*)d_out;   // [0,48): classifier logits; [48,176): gating probs
  float* out_logits = out;
  float* out_probs = out + B * 3;

  encoder_pass(C, gAq, gBq, gAv, gBv);
  k_gather_cls<<<(B * D + 255) / 256, 256, 0, stream>>>(w.x, w.clsb);
  mlp_head_run(C, w.gW1, gmb1, gmg, gmb, gmW2, gmb2, w.clsb, w.glog, E);
  k_gate_topk<<<1, 32, 0, stream>>>(w.glog, out_probs, w.w);

  k_zero<<<(B * D + 255) / 256, 256, 0, stream>>>(w.comb, B * D);
  for (int e = 0; e < E; ++e) {
    encoder_pass(C, eAq + (size_t)e * L * R * D, eBq + (size_t)e * L * D * R,
                    eAv + (size_t)e * L * R * D, eBv + (size_t)e * L * D * R);
    k_accum_cls<<<(B * D + 255) / 256, 256, 0, stream>>>(w.x, w.w, e, w.comb);
  }

  k_f32_to_bf16<<<(B * D + 255) / 256, 256, 0, stream>>>(w.comb, w.combb, B * D);
  mlp_head_run(C, w.cW1, cmb1, cmg, cmb, cmW2, cmb2, w.combb, out_logits, 3);
}
